// LiquidTimeConstantNetwork_85761906966564
// MI455X (gfx1250) — matrix operand order repaired and checked by reading, not compiled, not run on hardware
//
#include <hip/hip_runtime.h>
#include <cstdint>
#include <cstddef>

typedef __attribute__((ext_vector_type(16))) _Float16 v16h;
typedef __attribute__((ext_vector_type(8)))  float    v8f;
typedef _Float16 half_t;

#define VOCAB     32000
#define HID       256
#define SEQ       2048
#define BATCH     64
#define NSTEPS    4
#define STEP_SIZE 0.1f
#define TVC_MIN   1e-6f
#define TVC_MAX   1e6f
#define D_CLAMP   1e6f

// ---- LDS layout (bytes) ----
#define W_STRIDE   264                    // halfs per weight row (256+8 pad, 16B-aligned rows)
#define SZ_W     (HID * W_STRIDE * 2)     // 135168 per weight matrix
#define OFF_WR   0
#define OFF_WA   (OFF_WR + SZ_W)          // 135168
#define OFF_H16A (OFF_WA + SZ_W)          // 270336 (32B aligned)  h ping buffer
#define SZ_H16S  (8 * 32 * 16 * 2)        // 8192: [kb][lane][16 halfs] swizzled A-frag layout
#define OFF_H16B (OFF_H16A + SZ_H16S)     // 278528                h pong buffer
#define LDS_BYTES (OFF_H16B + SZ_H16S)    // 286720  (< 320KB WGP LDS)

// Fenced split barrier: all my LDS reads+writes retired, then signal; wait separately.
__device__ __forceinline__ void wg_signal() {
    asm volatile("s_wait_dscnt 0\n\ts_barrier_signal -1" ::: "memory");
}
__device__ __forceinline__ void wg_wait() {
    asm volatile("s_barrier_wait -1" ::: "memory");
}

// A fragment: swizzled layout -> one contiguous 32B read per lane (2x ds_load_b128).
__device__ __forceinline__ v16h load_a_frag(const half_t* h16s, int lane, int kb) {
    return *reinterpret_cast<const v16h*>(h16s + (kb * 32 + lane) * 16);
}

// B fragment (32x16 f16): lane holds column n, 16 contiguous K (lanes 16-31: K+16).
__device__ __forceinline__ v16h load_b_frag(const half_t* w, int n, int kglob) {
    union { v16h v; int4 q[2]; } b;
    const int4* sp = reinterpret_cast<const int4*>(w + n * W_STRIDE + kglob);
    b.q[0] = sp[0];
    b.q[1] = sp[1];
    return b.v;
}

extern "C" __global__ __launch_bounds__(512)
void ltc_fused_kernel(const int* __restrict__ tokens,
                      const float* __restrict__ emb,
                      const float* __restrict__ Wr,  const float* __restrict__ br,
                      const float* __restrict__ Wa,  const float* __restrict__ ba,
                      const float* __restrict__ tau,
                      const float* __restrict__ Wout, const float* __restrict__ bout,
                      float* __restrict__ out)
{
    extern __shared__ char lds[];
    half_t* wr16 = reinterpret_cast<half_t*>(lds + OFF_WR);
    half_t* wa16 = reinterpret_cast<half_t*>(lds + OFF_WA);
    half_t* h16a = reinterpret_cast<half_t*>(lds + OFF_H16A);

    const int tid  = threadIdx.x;
    const int lane = tid & 31;
    const int wave = tid >> 5;          // 0..15 -> N tile of 16 columns
    const int gb   = blockIdx.x * 16;   // batch row base

    // ---- Stage weights into LDS as f16 (resident for all 2048 steps) ----
    for (int i = tid; i < HID * HID; i += 512) {
        const int o = i >> 8, k = i & 255;
        wr16[o * W_STRIDE + k] = (half_t)Wr[i];
        wa16[o * W_STRIDE + k] = (half_t)Wa[i];
    }
    for (int i = tid; i < 2 * 8 * 32 * 16; i += 512)
        h16a[i] = (half_t)0.0f;         // zero both ping-pong buffers (contiguous)

    // Per-lane fixed coordinates + constants
    const int   ncol  = wave * 16 + (lane & 15);
    const float brv   = br[ncol];
    const float bav   = ba[ncol];
    const float tauv  = tau[ncol];
    const float boutv = bout[ncol];
    const int   mbase = (lane >> 4) * 8;   // C/D layout: m = v + mbase
    const int   kloc  = (lane >> 4) * 16;  // B layout K sub-offset

    // Swizzled-h16 write coordinates: value (m, ncol) -> h[(kbn*32 + m + 16*gA)*16 + jA]
    const int kbn = ncol >> 5;
    const int kp  = ncol & 31;
    const int gA  = (kp >> 4) & 1;
    const int kpp = kp & 15;                                   // {0..15}
    const int vA  = kpp >> 1;
    const int jA  = vA * 2 + (kpp & 1);
    const int hwr = (kbn * 32 + mbase + 16 * gA) * 16 + jA;    // + v*16 per v

    // ---- Wout B-fragments resident in VGPRs ----
    v16h wof[8];
#pragma unroll
    for (int kb = 0; kb < 8; ++kb) {
        const float* src = Wout + (size_t)ncol * HID + kb * 32 + kloc;
        v16h w;
#pragma unroll
        for (int j = 0; j < 16; ++j) w[j] = (half_t)src[j];
        wof[kb] = w;
    }

    // ---- h lives in registers: lane owns (mbase+v, ncol) for v=0..7 ----
    float hreg[8];
#pragma unroll
    for (int v = 0; v < 8; ++v) hreg[v] = 0.0f;

    // ---- register double-buffered embedding gather (lane-private columns) ----
    float xf[8];
#pragma unroll
    for (int v = 0; v < 8; ++v) {
        int tok = tokens[(size_t)(gb + mbase + v) * SEQ + 0];
        tok = tok < 0 ? 0 : (tok > (VOCAB - 1) ? (VOCAB - 1) : tok);
        xf[v] = emb[(size_t)tok * HID + ncol];
    }

    __syncthreads();   // weights + zeroed h visible

    for (int t = 0; t < SEQ; ++t) {
        // issue next step's gather now; completes under the substep compute
        const int tn = (t + 1 < SEQ) ? (t + 1) : t;
        float xn[8];
#pragma unroll
        for (int v = 0; v < 8; ++v) {
            int tok = tokens[(size_t)(gb + mbase + v) * SEQ + tn];
            tok = tok < 0 ? 0 : (tok > (VOCAB - 1) ? (VOCAB - 1) : tok);
            xn[v] = emb[(size_t)tok * HID + ncol];
        }

        // ---- 4 semi-implicit Euler sub-steps; h ping-pongs, one barrier each ----
        for (int s = 0; s < NSTEPS; ++s) {
            const half_t* hr = h16a + (s & 1) * (8 * 32 * 16);        // read buffer
            half_t*       hw = h16a + ((s + 1) & 1) * (8 * 32 * 16);  // write buffer

            v8f aR0 = {}, aR1 = {}, aA0 = {}, aA1 = {};   // 2 chains per matmul
            for (int kb = 0; kb < 4; ++kb) {
                const int k0 = 2 * kb, k1 = 2 * kb + 1;
                const v16h a0  = load_a_frag(hr, lane, k0);
                const v16h a1  = load_a_frag(hr, lane, k1);
                const v16h bR0 = load_b_frag(wr16, ncol, k0 * 32 + kloc);
                const v16h bA0 = load_b_frag(wa16, ncol, k0 * 32 + kloc);
                const v16h bR1 = load_b_frag(wr16, ncol, k1 * 32 + kloc);
                const v16h bA1 = load_b_frag(wa16, ncol, k1 * 32 + kloc);
                aR0 = __builtin_amdgcn_wmma_f32_16x16x32_f16(false, a0, false, bR0, (short)0, aR0, false, false);
                aA0 = __builtin_amdgcn_wmma_f32_16x16x32_f16(false, a0, false, bA0, (short)0, aA0, false, false);
                aR1 = __builtin_amdgcn_wmma_f32_16x16x32_f16(false, a1, false, bR1, (short)0, aR1, false, false);
                aA1 = __builtin_amdgcn_wmma_f32_16x16x32_f16(false, a1, false, bA1, (short)0, aA1, false, false);
            }

            // elementwise update: pure registers (h in hreg, x in xf)
#pragma unroll
            for (int v = 0; v < 8; ++v) {
                const float hv = hreg[v];
                const float sr = xf[v] + (aR0[v] + aR1[v]) + brv;   // S_
                const float sa = sr + (aA0[v] + aA1[v]) + bav;      // tanh arg
                float tv = tauv * tanhf(sa);
                tv = fminf(fmaxf(tv, TVC_MIN), TVC_MAX);
                float dv = (sr - hv) / tv;
                dv = fminf(fmaxf(dv, -D_CLAMP), D_CLAMP);
                const float hn = hv + STEP_SIZE * dv;
                hreg[v] = hn;
                hw[hwr + v * 16] = (half_t)hn;       // publish to pong buffer
            }

            wg_signal();   // reads of hr + writes to hw retired (s_wait_dscnt 0)
            wg_wait();     // new h visible to all waves
        }

        // ---- fused output projection: y_t = h_t @ Wout^T + bout (h in buf0) ----
        v8f o0 = {}, o1 = {};
        for (int kb = 0; kb < 4; ++kb) {
            const v16h a0 = load_a_frag(h16a, lane, 2 * kb);
            const v16h a1 = load_a_frag(h16a, lane, 2 * kb + 1);
            o0 = __builtin_amdgcn_wmma_f32_16x16x32_f16(false, a0, false, wof[2 * kb],     (short)0, o0, false, false);
            o1 = __builtin_amdgcn_wmma_f32_16x16x32_f16(false, a1, false, wof[2 * kb + 1], (short)0, o1, false, false);
        }
#pragma unroll
        for (int v = 0; v < 8; ++v) {
            const int m = mbase + v;
            out[((size_t)(gb + m) * SEQ + t) * HID + ncol] = (o0[v] + o1[v]) + boutv;
        }

        // rotate gather double-buffer
#pragma unroll
        for (int v = 0; v < 8; ++v) xf[v] = xn[v];
    }
}

extern "C" void kernel_launch(void* const* d_in, const int* in_sizes, int n_in,
                              void* d_out, int out_size, void* d_ws, size_t ws_size,
                              hipStream_t stream) {
    (void)in_sizes; (void)n_in; (void)out_size; (void)d_ws; (void)ws_size;
    const int*   tokens = (const int*)d_in[0];
    const float* emb    = (const float*)d_in[1];
    const float* Wr     = (const float*)d_in[2];
    const float* brp    = (const float*)d_in[3];
    const float* Wa     = (const float*)d_in[4];
    const float* bap    = (const float*)d_in[5];
    const float* taup   = (const float*)d_in[6];
    const float* Wout   = (const float*)d_in[7];
    const float* boutp  = (const float*)d_in[8];
    float* out = (float*)d_out;

    (void)hipFuncSetAttribute(reinterpret_cast<const void*>(ltc_fused_kernel),
                              hipFuncAttributeMaxDynamicSharedMemorySize, LDS_BYTES);

    dim3 grid(BATCH / 16);   // 4 workgroups, one per 16 batch rows
    dim3 block(512);         // 16 waves (wave32), one N-tile each
    ltc_fused_kernel<<<grid, block, LDS_BYTES, stream>>>(
        tokens, emb, Wr, brp, Wa, bap, taup, Wout, boutp, out);
}
